// GlobalAttention_86131274154404
// MI455X (gfx1250) — matrix operand order repaired and checked by reading, not compiled, not run on hardware
//
#include <hip/hip_runtime.h>
#include <hip/hip_bf16.h>

typedef __attribute__((ext_vector_type(16))) __bf16 v16bf;
typedef __attribute__((ext_vector_type(8)))  __bf16 v8bf;
typedef __attribute__((ext_vector_type(8)))  float  v8f;

// ---------- helpers ----------
__device__ __forceinline__ __bf16 f2bf(float f) {
    unsigned u = __builtin_bit_cast(unsigned, f);
    u += 0x7fffu + ((u >> 16) & 1u);                 // round-to-nearest-even
    unsigned short h = (unsigned short)(u >> 16);
    return __builtin_bit_cast(__bf16, h);
}

__device__ __forceinline__ v16bf combine8(v8bf lo, v8bf hi) {
    return __builtin_shufflevector(lo, hi, 0,1,2,3,4,5,6,7,8,9,10,11,12,13,14,15);
}

__device__ __forceinline__ v8f wmma_bf16(v16bf a, v16bf b, v8f c) {
    return __builtin_amdgcn_wmma_f32_16x16x32_bf16(false, a, false, b, (short)0, c,
                                                   false, false);
}

// Load A fragment (16x32 bf16): lane holds row, two contiguous 8-elem chunks.
__device__ __forceinline__ v16bf load_afrag(const __bf16* row, int k0, int lh) {
    v8bf lo = *(const v8bf*)(row + k0 + 16 * lh);
    v8bf hi = *(const v8bf*)(row + k0 + 16 * lh + 8);
    return combine8(lo, hi);
}

// ---------- prep kernels ----------
// im2col (stride==kernel==7, so a pure relayout) + fp32->bf16.
// xp[((n*12+ox)*12+oy)][ (ky*7+kx)*256 + c ] = x[n][7ox+ky][7oy+kx][c]
__global__ void k_im2col(const float* __restrict__ x, __bf16* __restrict__ xp) {
    int idx = blockIdx.x * blockDim.x + threadIdx.x;
    if (idx >= 1152 * 12544) return;
    int r = idx / 12544, p = idx % 12544;
    int c  = p % 256;
    int kx = (p / 256) % 7;
    int ky =  p / (256 * 7);
    int oy = r % 12, ox = (r / 12) % 12, n = r / 144;
    int X = ox * 7 + ky, Y = oy * 7 + kx;
    xp[idx] = f2bf(x[((size_t)(n * 84 + X) * 84 + Y) * 256 + c]);
}

// dst[c*R + r] = src[r*C + c]  (weights -> N-major bf16)
__global__ void k_tcvt(const float* __restrict__ src, __bf16* __restrict__ dst,
                       int R, int C) {
    int idx = blockIdx.x * blockDim.x + threadIdx.x;
    if (idx >= R * C) return;
    int r = idx / C, c = idx % C;
    dst[(size_t)c * R + r] = f2bf(src[idx]);
}

// vT[bh][d][j(padded to 160)] = v_j[d]  (v = kv channels 512..1023, head-sliced)
__global__ void k_build_vT(const __bf16* __restrict__ kvb, __bf16* __restrict__ vT) {
    int idx = blockIdx.x * blockDim.x + threadIdx.x;
    if (idx >= 64 * 64 * 160) return;
    int j  = idx % 160;
    int d  = (idx / 160) % 64;
    int bh = idx / (160 * 64);
    int n = bh >> 3, h = bh & 7;
    __bf16 val = __builtin_bit_cast(__bf16, (unsigned short)0);
    if (j < 144) val = kvb[(size_t)(n * 144 + j) * 1024 + 512 + h * 64 + d];
    vT[idx] = val;
}

// ---------- GEMM kernels ----------
// 4 waves/block; each wave owns a 16x64 C strip (4 tiles share one A fragment
// per 32-k chunk -> 4 wmma : 10 b128 loads instead of 1 : 4).

// q = x @ w_q : M=56448 K=256 N=512 ; A rows come from the im2col buffer remapped.
__global__ __launch_bounds__(128) void k_gemm_q(const __bf16* __restrict__ xp,
                                                const __bf16* __restrict__ wqt,
                                                __bf16* __restrict__ qb) {
    const int lane = threadIdx.x & 31, wave = threadIdx.x >> 5;
    const int lmod = lane & 15, lh = lane >> 4;
    const int row0 = blockIdx.x * 64 + wave * 16;
    const int col0 = blockIdx.y * 64;
    int g = row0 + lmod;                     // pixel id for this lane's A row
    int n = g / 7056, rem = g % 7056;
    int X = rem / 84, Y = rem % 84;
    const __bf16* arow =
        xp + (size_t)(((n * 12 + X / 7) * 12 + Y / 7) * 49 + (X % 7) * 7 + (Y % 7)) * 256;
    const __bf16* brow = wqt + (size_t)(col0 + lmod) * 256;   // + t*16*256 per tile
    v8f acc[4] = {{}, {}, {}, {}};
#pragma unroll
    for (int k0 = 0; k0 < 256; k0 += 32) {
        v16bf a = load_afrag(arow, k0, lh);
#pragma unroll
        for (int t = 0; t < 4; ++t) {
            v16bf b = *(const v16bf*)(brow + (size_t)t * 16 * 256 + k0 + 16 * lh);
            acc[t] = wmma_bf16(a, b, acc[t]);
        }
    }
#pragma unroll
    for (int t = 0; t < 4; ++t)
#pragma unroll
        for (int r = 0; r < 8; ++r)
            qb[(size_t)(row0 + r + 8 * lh) * 512 + col0 + t * 16 + lmod] = f2bf(acc[t][r]);
}

// kv = patches @ w_kv : M=1152 K=12544 N=1024
__global__ __launch_bounds__(128) void k_gemm_kv(const __bf16* __restrict__ xp,
                                                 const __bf16* __restrict__ wkvt,
                                                 __bf16* __restrict__ kvb) {
    const int lane = threadIdx.x & 31, wave = threadIdx.x >> 5;
    const int lmod = lane & 15, lh = lane >> 4;
    const int row0 = blockIdx.x * 64 + wave * 16;
    const int col0 = blockIdx.y * 64;
    const __bf16* arow = xp   + (size_t)(row0 + lmod) * 12544;
    const __bf16* brow = wkvt + (size_t)(col0 + lmod) * 12544;
    v8f acc[4] = {{}, {}, {}, {}};
#pragma unroll 2
    for (int k0 = 0; k0 < 12544; k0 += 32) {
        v16bf a = load_afrag(arow, k0, lh);
#pragma unroll
        for (int t = 0; t < 4; ++t) {
            v16bf b = *(const v16bf*)(brow + (size_t)t * 16 * 12544 + k0 + 16 * lh);
            acc[t] = wmma_bf16(a, b, acc[t]);
        }
    }
#pragma unroll
    for (int t = 0; t < 4; ++t)
#pragma unroll
        for (int r = 0; r < 8; ++r)
            kvb[(size_t)(row0 + r + 8 * lh) * 1024 + col0 + t * 16 + lmod] = f2bf(acc[t][r]);
}

// attention: one wave handles 16 q rows of one batch-head.
__global__ __launch_bounds__(32) void k_attn(const __bf16* __restrict__ qb,
                                             const __bf16* __restrict__ kvb,
                                             const __bf16* __restrict__ vT,
                                             __bf16* __restrict__ oa) {
    __shared__ float probs[16][168];         // 144 scores + zero pad to 160
    const int lane = threadIdx.x & 31;
    const int lmod = lane & 15, lh = lane >> 4;
    const int bh = blockIdx.y;
    const int n = bh >> 3, h = bh & 7;
    const size_t g0 = (size_t)n * 7056 + blockIdx.x * 16;

    // q fragments (K=64 -> two 16x32 A frags), rows contiguous in qb
    const __bf16* arow = qb + (size_t)(g0 + lmod) * 512 + h * 64;
    v16bf a0 = load_afrag(arow, 0, lh);
    v16bf a1 = load_afrag(arow, 32, lh);

    // scores = q @ k^T * scale ; B^T rows are the k-vectors themselves
#pragma unroll
    for (int t = 0; t < 9; ++t) {
        const __bf16* brow = kvb + (size_t)(n * 144 + t * 16 + lmod) * 1024 + h * 64;
        v16bf b0 = *(const v16bf*)(brow + 16 * lh);
        v16bf b1 = *(const v16bf*)(brow + 32 + 16 * lh);
        v8f s = {};
        s = wmma_bf16(a0, b0, s);
        s = wmma_bf16(a1, b1, s);
#pragma unroll
        for (int r = 0; r < 8; ++r)
            probs[r + 8 * lh][t * 16 + lmod] = s[r] * 0.125f;   // SCALE = 64^-0.5
    }
#pragma unroll
    for (int i = 0; i < 8; ++i) probs[lmod][144 + 8 * lh + i] = 0.0f;
    __syncthreads();

    if (lane < 16) {                         // row-wise softmax over 144
        float mx = -1e30f;
        for (int c = 0; c < 144; ++c) mx = fmaxf(mx, probs[lane][c]);
        float sum = 0.0f;
        for (int c = 0; c < 144; ++c) {
            float e = __expf(probs[lane][c] - mx);
            probs[lane][c] = e;
            sum += e;
        }
        float inv = 1.0f / sum;
        for (int c = 0; c < 144; ++c) probs[lane][c] *= inv;
    }
    __syncthreads();                         // re-converged: EXEC all-ones for WMMA

    // out = attn(16x160) @ v(160x64), vT is N-major and zero-padded
#pragma unroll
    for (int nt = 0; nt < 4; ++nt) {
        const int d0 = nt * 16;
        v8f acc = {};
#pragma unroll
        for (int kc = 0; kc < 5; ++kc) {
            const int k0 = kc * 32;
            v16bf a;
#pragma unroll
            for (int e = 0; e < 16; ++e) {
                int k = k0 + 16 * lh + e;  // A-frag k mapping
                a[e] = f2bf(probs[lmod][k]);
            }
            v16bf b = *(const v16bf*)(vT + ((size_t)bh * 64 + d0 + lmod) * 160 + k0 + 16 * lh);
            acc = wmma_bf16(a, b, acc);
        }
#pragma unroll
        for (int r = 0; r < 8; ++r)
            oa[(g0 + r + 8 * lh) * 512 + h * 64 + d0 + lmod] = f2bf(acc[r]);
    }
}

// out = attn_out @ w_out + b : M=56448 K=512 N=256, fp32 output
__global__ __launch_bounds__(128) void k_gemm_out(const __bf16* __restrict__ oa,
                                                  const __bf16* __restrict__ wot,
                                                  const float* __restrict__ bias,
                                                  float* __restrict__ out) {
    const int lane = threadIdx.x & 31, wave = threadIdx.x >> 5;
    const int lmod = lane & 15, lh = lane >> 4;
    const int row0 = blockIdx.x * 64 + wave * 16;
    const int col0 = blockIdx.y * 64;
    const __bf16* arow = oa  + (size_t)(row0 + lmod) * 512;
    const __bf16* brow = wot + (size_t)(col0 + lmod) * 512;
    v8f acc[4] = {{}, {}, {}, {}};
#pragma unroll
    for (int k0 = 0; k0 < 512; k0 += 32) {
        v16bf a = load_afrag(arow, k0, lh);
#pragma unroll
        for (int t = 0; t < 4; ++t) {
            v16bf b = *(const v16bf*)(brow + (size_t)t * 16 * 512 + k0 + 16 * lh);
            acc[t] = wmma_bf16(a, b, acc[t]);
        }
    }
#pragma unroll
    for (int t = 0; t < 4; ++t) {
        float bv = bias[col0 + t * 16 + lmod];
#pragma unroll
        for (int r = 0; r < 8; ++r)
            out[(size_t)(row0 + r + 8 * lh) * 256 + col0 + t * 16 + lmod] = acc[t][r] + bv;
    }
}

// ---------- host ----------
extern "C" void kernel_launch(void* const* d_in, const int* in_sizes, int n_in,
                              void* d_out, int out_size, void* d_ws, size_t ws_size,
                              hipStream_t stream) {
    const float* x    = (const float*)d_in[0];
    const float* wq   = (const float*)d_in[1];
    const float* wkv  = (const float*)d_in[2];
    const float* wout = (const float*)d_in[3];
    const float* bout = (const float*)d_in[4];
    float* out = (float*)d_out;

    char* ws = (char*)d_ws;
    size_t off = 0;
    auto carve = [&](size_t bytes) -> char* {
        off = (off + 255) & ~(size_t)255;
        char* p = ws + off;
        off += bytes;
        return p;
    };
    __bf16* xp   = (__bf16*)carve((size_t)1152 * 12544 * 2);   // im2col patches
    __bf16* wqt  = (__bf16*)carve((size_t)512 * 256 * 2);      // w_q^T
    __bf16* wkvt = (__bf16*)carve((size_t)1024 * 12544 * 2);   // w_kv^T
    __bf16* wot  = (__bf16*)carve((size_t)256 * 512 * 2);      // w_out^T
    __bf16* qb   = (__bf16*)carve((size_t)56448 * 512 * 2);    // q (bf16)
    __bf16* kvb  = (__bf16*)carve((size_t)1152 * 1024 * 2);    // k|v (bf16)
    __bf16* vT   = (__bf16*)carve((size_t)64 * 64 * 160 * 2);  // v transposed+padded
    __bf16* oa   = (__bf16*)carve((size_t)56448 * 512 * 2);    // attention output

    k_im2col<<<(1152 * 12544 + 255) / 256, 256, 0, stream>>>(x, xp);
    k_tcvt<<<(256 * 512 + 255) / 256, 256, 0, stream>>>(wq, wqt, 256, 512);
    k_tcvt<<<(12544 * 1024 + 255) / 256, 256, 0, stream>>>(wkv, wkvt, 12544, 1024);
    k_tcvt<<<(512 * 256 + 255) / 256, 256, 0, stream>>>(wout, wot, 512, 256);

    k_gemm_q<<<dim3(56448 / 64, 512 / 64), 128, 0, stream>>>(xp, wqt, qb);
    k_gemm_kv<<<dim3(1152 / 64, 1024 / 64), 128, 0, stream>>>(xp, wkvt, kvb);
    k_build_vT<<<(64 * 64 * 160 + 255) / 256, 256, 0, stream>>>(kvb, vT);

    k_attn<<<dim3(7056 / 16, 64), 32, 0, stream>>>(qb, kvb, vT, oa);

    k_gemm_out<<<dim3(56448 / 64, 256 / 64), 128, 0, stream>>>(oa, wot, bout, out);
}
